// CausalSelfAttention_30880814858322
// MI455X (gfx1250) — matrix operand order repaired and checked by reading, not compiled, not run on hardware
//
#include <hip/hip_runtime.h>
#include <hip/hip_bf16.h>
#include <stdint.h>

// Problem constants (from reference): B=4, T=2048, C=1024, H=16, d=64
#define BB    4
#define TT    2048
#define CC    1024
#define HH    16
#define DD    64
#define MTOT  (BB*TT)    // 8192 token rows
#define NQKV  (3*CC)     // 3072

typedef __attribute__((ext_vector_type(16))) __bf16 bf16x16;
typedef __attribute__((ext_vector_type(8)))  __bf16 bf16x8;
typedef __attribute__((ext_vector_type(8)))  float  f32x8;

__device__ __forceinline__ bf16x16 cat8(bf16x8 lo, bf16x8 hi) {
  return __builtin_shufflevector(lo, hi, 0,1,2,3,4,5,6,7,8,9,10,11,12,13,14,15);
}
__device__ __forceinline__ f32x8 wmma_bf16(bf16x16 a, bf16x16 b, f32x8 c) {
  // (neg_a, A, neg_b, B, c_mod, C, reuse_a, reuse_b)
  return __builtin_amdgcn_wmma_f32_16x16x32_bf16(false, a, false, b, (short)0, c, false, false);
}
__device__ __forceinline__ f32x8 zero8() {
  f32x8 v;
#pragma unroll
  for (int i = 0; i < 8; ++i) v[i] = 0.0f;
  return v;
}

// ---------------------------------------------------------------------------
// Precision conversion kernels
// ---------------------------------------------------------------------------
__global__ __launch_bounds__(256)
void cvt_f32_to_bf16(const float* __restrict__ in, __bf16* __restrict__ out, int n) {
  int i = blockIdx.x * blockDim.x + threadIdx.x;
  if (i < n) out[i] = (__bf16)in[i];
}

// in:  [K=1024][N] f32 (x @ W convention)  ->  out: [N][1024] bf16 (B^T layout)
__global__ __launch_bounds__(256)
void transpose_w_to_bf16(const float* __restrict__ in, __bf16* __restrict__ out, int N) {
  int i = blockIdx.x * blockDim.x + threadIdx.x;
  int total = N * CC;
  if (i < total) {
    int n = i / CC;
    int k = i - n * CC;
    out[i] = (__bf16)in[(size_t)k * N + n];
  }
}

// ---------------------------------------------------------------------------
// Shared bf16 GEMM mainloop: acc[2][4] = A[64 x 1024] * Bt[N x 1024]^T tile
// Block: 256 threads = 8 waves, block tile M=64, N=256.
// wave (wm, wn): wm = wid>>2 (2 M-subtiles of 16), wn = wid&3 (4 N-subtiles of 16)
// ---------------------------------------------------------------------------
__device__ __forceinline__ void gemm_mainloop(const __bf16* __restrict__ A,
                                              const __bf16* __restrict__ Bt,
                                              int mBase, int nBase,
                                              int wave_m, int wave_n,
                                              f32x8 acc[2][4]) {
  __shared__ __bf16 Alds[64 * 32];    // 4 KB   row stride 32 elems
  __shared__ __bf16 Blds[256 * 32];   // 16 KB  row stride 32 elems ([N][K] tile)

  const int tid  = threadIdx.x;
  const int lane = tid & 31;
  const int l16  = lane & 15;
  const int half = lane >> 4;

#pragma unroll
  for (int mi = 0; mi < 2; ++mi)
#pragma unroll
    for (int nj = 0; nj < 4; ++nj) acc[mi][nj] = zero8();

  const int ldRow = tid >> 2;         // 0..63
  const int ldCh  = tid & 3;          // 0..3 (16B chunks of the 64B K-row)

  for (int k0 = 0; k0 < CC; k0 += 32) {
    __syncthreads();
    // Stage A tile (64 rows x 32 K) -- one b128 per thread
    {
      const bf16x8* src =
          (const bf16x8*)(A + (size_t)(mBase + ldRow) * CC + k0 + ldCh * 8);
      *(bf16x8*)(Alds + ldRow * 32 + ldCh * 8) = *src;
      __builtin_prefetch(A + (size_t)(mBase + ldRow) * CC + k0 + 32 + ldCh * 8, 0, 0);
    }
    // Stage B tile (256 N-rows x 32 K) -- four b128 per thread
#pragma unroll
    for (int i = 0; i < 4; ++i) {
      int nrow = ldRow + i * 64;
      const bf16x8* src =
          (const bf16x8*)(Bt + (size_t)(nBase + nrow) * CC + k0 + ldCh * 8);
      *(bf16x8*)(Blds + nrow * 32 + ldCh * 8) = *src;
    }
    __syncthreads();

    // Gather A fragments (ISA 16-bit A layout: elems 0..7 -> K=half*8+e,
    // elems 8..15 -> K=16+half*8+e)
    bf16x16 af[2];
#pragma unroll
    for (int mi = 0; mi < 2; ++mi) {
      int row = wave_m * 32 + mi * 16 + l16;
      bf16x8 lo = *(const bf16x8*)(Alds + row * 32 + half * 8);
      bf16x8 hi = *(const bf16x8*)(Alds + row * 32 + 16 + half * 8);
      af[mi] = cat8(lo, hi);
    }
    // B fragments (ISA B layout: lane holds N=l16 column, elems 0..7 -> K=half*8+e, elems 8..15 -> K=16+half*8+e)
#pragma unroll
    for (int nj = 0; nj < 4; ++nj) {
      int nrow = wave_n * 64 + nj * 16 + l16;
      bf16x8 lo = *(const bf16x8*)(Blds + nrow * 32 + half * 8);
      bf16x8 hi = *(const bf16x8*)(Blds + nrow * 32 + 16 + half * 8);
      bf16x16 bf = cat8(lo, hi);
#pragma unroll
      for (int mi = 0; mi < 2; ++mi)
        acc[mi][nj] = wmma_bf16(af[mi], bf, acc[mi][nj]);
    }
  }
}

// ---------------------------------------------------------------------------
// GEMM 1: qkv = xb @ Wqkv ; epilogue scatters Q,K as [B,H,T,d] bf16 and
// V transposed as [B,H,d,T] bf16 (so PV B-fragments are K-contiguous).
// part (q/k/v) and head h are uniform per 16-wide N-subtile, and the batch
// index b is uniform per block (64-aligned M tiles never straddle T=2048),
// so all selection logic is scalar (SGPR) -> uniform branches, no per-lane
// branch nest.
// ---------------------------------------------------------------------------
__global__ __launch_bounds__(256)
void gemm_qkv(const __bf16* __restrict__ xb, const __bf16* __restrict__ Wt,
              __bf16* __restrict__ Qb, __bf16* __restrict__ Kb,
              __bf16* __restrict__ Vt) {
  const int mBase = blockIdx.x * 64;
  const int nBase = blockIdx.y * 256;
  const int tid  = threadIdx.x;
  const int lane = tid & 31;
  const int l16  = lane & 15;
  const int half = lane >> 4;
  // wave coords are wave-uniform: force into SGPRs
  const int wid    = __builtin_amdgcn_readfirstlane(tid >> 5);
  const int wave_m = wid >> 2, wave_n = wid & 3;

  f32x8 acc[2][4];
  gemm_mainloop(xb, Wt, mBase, nBase, wave_m, wave_n, acc);

  const int b  = mBase >> 11;          // scalar: batch index (block-uniform)
  const int t0 = (mBase & 2047) + wave_m * 32;  // scalar base token-in-batch

#pragma unroll
  for (int nj = 0; nj < 4; ++nj) {
    const int ntile = nBase + wave_n * 64 + nj * 16;  // scalar tile base col
    const int part  = ntile >> 10;                    // 0:q 1:k 2:v (scalar)
    const int cc0   = ntile & 1023;
    const int h     = cc0 >> 6;                       // head (scalar)
    const int dd    = (cc0 & 63) + l16;               // head-dim element
    const size_t bh = (size_t)b * HH + h;
    if (part < 2) {
      __bf16* __restrict__ dst = (part == 0) ? Qb : Kb;
      __bf16* __restrict__ base = dst + bh * TT * DD + dd;
#pragma unroll
      for (int mi = 0; mi < 2; ++mi) {
#pragma unroll
        for (int v = 0; v < 8; ++v) {
          int t = t0 + mi * 16 + v + 8 * half;
          base[(size_t)t * DD] = (__bf16)acc[mi][nj][v];
        }
      }
    } else {
      __bf16* __restrict__ base = Vt + (bh * DD + dd) * TT;
#pragma unroll
      for (int mi = 0; mi < 2; ++mi) {
#pragma unroll
        for (int v = 0; v < 8; ++v) {
          int t = t0 + mi * 16 + v + 8 * half;
          base[t] = (__bf16)acc[mi][nj][v];
        }
      }
    }
  }
}

// ---------------------------------------------------------------------------
// Flash attention: grid (T/64, B*H), 128 threads = 4 waves.
// Wave w owns query rows qt*64 + w*16 .. +15; online softmax; WMMA for
// S = Q K^T (K-dim = d = 64) and O += P V (K-dim = key-tile = 64).
// ---------------------------------------------------------------------------
__global__ __launch_bounds__(128)
void attn_fwd(const __bf16* __restrict__ Qb, const __bf16* __restrict__ Kb,
              const __bf16* __restrict__ Vt, __bf16* __restrict__ yb) {
  __shared__ __bf16 Plds[4 * 16 * 64];   // 2 KB per wave

  const int tid  = threadIdx.x;
  const int lane = tid & 31;
  const int w    = __builtin_amdgcn_readfirstlane(tid >> 5);  // scalar wave id
  const int l16  = lane & 15;
  const int half = lane >> 4;

  const int qt = blockIdx.x;
  const int bh = blockIdx.y;
  const int b  = bh >> 4;
  const int h  = bh & 15;
  const int qbase = qt * 64 + w * 16;    // scalar

  const __bf16* Q  = Qb + (size_t)bh * TT * DD;
  const __bf16* K  = Kb + (size_t)bh * TT * DD;
  const __bf16* V  = Vt + (size_t)bh * DD * TT;
  __bf16*       Pw = Plds + w * 16 * 64;

  // Q A-fragments, kept in registers for the whole row-block (d=64 -> 2 chunks)
  bf16x16 qf[2];
  {
    const __bf16* qrow = Q + (size_t)(qbase + l16) * DD;
#pragma unroll
    for (int c = 0; c < 2; ++c) {
      bf16x8 lo = *(const bf16x8*)(qrow + c * 32 + half * 8);
      bf16x8 hi = *(const bf16x8*)(qrow + c * 32 + 16 + half * 8);
      qf[c] = cat8(lo, hi);
    }
  }

  f32x8 o[4];
#pragma unroll
  for (int dt = 0; dt < 4; ++dt) o[dt] = zero8();
  float mrow[8], lrow[8];
#pragma unroll
  for (int v = 0; v < 8; ++v) { mrow[v] = -INFINITY; lrow[v] = 0.0f; }

  const float scale = 0.125f;   // 1/sqrt(64)
  const int kb_end = qt * 64;   // causal: last key tile starts at query tile

  for (int kb = 0; kb <= kb_end; kb += 64) {
    // ---- S = Q K^T for this 16 x 64 strip ----
    f32x8 s[4];
#pragma unroll
    for (int nt = 0; nt < 4; ++nt) {
      s[nt] = zero8();
#pragma unroll
      for (int c = 0; c < 2; ++c) {
        const __bf16* kr =
            K + (size_t)(kb + nt * 16 + l16) * DD + c * 32 + half * 8;
        bf16x16 kf = cat8(*(const bf16x8*)kr, *(const bf16x8*)(kr + 16));
        s[nt] = wmma_bf16(qf[c], kf, s[nt]);
      }
    }

    // ---- scale + causal mask + tile row-max ----
    float tmax[8];
#pragma unroll
    for (int v = 0; v < 8; ++v) tmax[v] = -INFINITY;
#pragma unroll
    for (int nt = 0; nt < 4; ++nt) {
#pragma unroll
      for (int v = 0; v < 8; ++v) {
        int col = kb + nt * 16 + l16;
        int row = qbase + v + 8 * half;
        float val = (col <= row) ? s[nt][v] * scale : -INFINITY;
        s[nt][v] = val;
        tmax[v] = fmaxf(tmax[v], val);
      }
    }
    // D-layout: a row's 16 N-values live in one 16-lane half -> xor reduce
#pragma unroll
    for (int v = 0; v < 8; ++v) {
#pragma unroll
      for (int off = 1; off < 16; off <<= 1)
        tmax[v] = fmaxf(tmax[v], __shfl_xor(tmax[v], off));
    }

    float sc[8], rs[8];
#pragma unroll
    for (int v = 0; v < 8; ++v) {
      float nm = fmaxf(mrow[v], tmax[v]);
      sc[v] = __expf(mrow[v] - nm);
      mrow[v] = nm;
      rs[v] = 0.0f;
    }

    // ---- P = exp(S - m), stash to LDS (D-layout -> memory), row sums ----
#pragma unroll
    for (int nt = 0; nt < 4; ++nt) {
#pragma unroll
      for (int v = 0; v < 8; ++v) {
        float p = __expf(s[nt][v] - mrow[v]);
        rs[v] += p;
        Pw[(v + 8 * half) * 64 + nt * 16 + l16] = (__bf16)p;
      }
    }
#pragma unroll
    for (int v = 0; v < 8; ++v) {
#pragma unroll
      for (int off = 1; off < 16; off <<= 1)
        rs[v] += __shfl_xor(rs[v], off);
      lrow[v] = lrow[v] * sc[v] + rs[v];
    }
#pragma unroll
    for (int dt = 0; dt < 4; ++dt)
#pragma unroll
      for (int v = 0; v < 8; ++v) o[dt][v] *= sc[v];

    // wave-local LDS RAW: stores above -> A-layout loads below
    asm volatile("s_wait_dscnt 0" ::: "memory");

    // ---- P A-fragments (A layout from LDS) ----
    bf16x16 pf[2];
#pragma unroll
    for (int c = 0; c < 2; ++c) {
      const __bf16* pr = Pw + l16 * 64 + c * 32 + half * 8;
      pf[c] = cat8(*(const bf16x8*)pr, *(const bf16x8*)(pr + 16));
    }

    // ---- O += P V : V stored [d][T] so B-fragments are contiguous ----
#pragma unroll
    for (int dt = 0; dt < 4; ++dt) {
#pragma unroll
      for (int c = 0; c < 2; ++c) {
        const __bf16* vr =
            V + (size_t)(dt * 16 + l16) * TT + kb + c * 32 + half * 8;
        bf16x16 vf = cat8(*(const bf16x8*)vr, *(const bf16x8*)(vr + 16));
        o[dt] = wmma_bf16(pf[c], vf, o[dt]);
      }
    }
  }

  // ---- normalize and write y[B,T,C] bf16 ----
  __bf16* yrow0 = yb + (size_t)b * TT * CC + (size_t)qbase * CC + h * 64 + l16;
#pragma unroll
  for (int dt = 0; dt < 4; ++dt) {
#pragma unroll
    for (int v = 0; v < 8; ++v) {
      int r = v + 8 * half;
      float val = o[dt][v] / lrow[v];
      yrow0[(size_t)r * CC + dt * 16] = (__bf16)val;
    }
  }
}

// ---------------------------------------------------------------------------
// GEMM 2: out = yb @ W_out, f32 epilogue straight to d_out
// ---------------------------------------------------------------------------
__global__ __launch_bounds__(256)
void gemm_out(const __bf16* __restrict__ yb, const __bf16* __restrict__ Wt,
              float* __restrict__ out) {
  const int mBase = blockIdx.x * 64;
  const int nBase = blockIdx.y * 256;
  const int tid  = threadIdx.x;
  const int lane = tid & 31;
  const int l16  = lane & 15;
  const int half = lane >> 4;
  const int wid    = __builtin_amdgcn_readfirstlane(tid >> 5);
  const int wave_m = wid >> 2, wave_n = wid & 3;

  f32x8 acc[2][4];
  gemm_mainloop(yb, Wt, mBase, nBase, wave_m, wave_n, acc);

  float* base = out + (size_t)(mBase + wave_m * 32 + 8 * half) * CC +
                nBase + wave_n * 64 + l16;
#pragma unroll
  for (int mi = 0; mi < 2; ++mi) {
#pragma unroll
    for (int nj = 0; nj < 4; ++nj) {
#pragma unroll
      for (int v = 0; v < 8; ++v) {
        base[(size_t)(mi * 16 + v) * CC + nj * 16] = acc[mi][nj][v];
      }
    }
  }
}

// ---------------------------------------------------------------------------
// Host launcher
// ---------------------------------------------------------------------------
extern "C" void kernel_launch(void* const* d_in, const int* in_sizes, int n_in,
                              void* d_out, int out_size, void* d_ws, size_t ws_size,
                              hipStream_t stream) {
  const float* x     = (const float*)d_in[0];   // [4,2048,1024]
  const float* W_qkv = (const float*)d_in[1];   // [1024,3072]
  const float* W_out = (const float*)d_in[2];   // [1024,1024]
  float* out = (float*)d_out;                   // [4,2048,1024]

  // Workspace carve-up (all 256B aligned sizes)
  uint8_t* ws = (uint8_t*)d_ws;
  size_t off = 0;
  __bf16* xb   = (__bf16*)(ws + off); off += (size_t)MTOT * CC * 2;         // 16 MB
  __bf16* wqkt = (__bf16*)(ws + off); off += (size_t)NQKV * CC * 2;         // 6 MB
  __bf16* wot  = (__bf16*)(ws + off); off += (size_t)CC * CC * 2;           // 2 MB
  __bf16* Qb   = (__bf16*)(ws + off); off += (size_t)BB * HH * TT * DD * 2; // 16 MB
  __bf16* Kb   = (__bf16*)(ws + off); off += (size_t)BB * HH * TT * DD * 2; // 16 MB
  __bf16* Vt   = (__bf16*)(ws + off); off += (size_t)BB * HH * TT * DD * 2; // 16 MB
  __bf16* yb   = (__bf16*)(ws + off); off += (size_t)MTOT * CC * 2;         // 16 MB
  if (ws_size < off) return;  // workspace too small: do nothing (deterministic)

  // 1) precision conversion / weight transposition
  {
    int n = MTOT * CC;
    cvt_f32_to_bf16<<<(n + 255) / 256, 256, 0, stream>>>(x, xb, n);
  }
  {
    int n = NQKV * CC;
    transpose_w_to_bf16<<<(n + 255) / 256, 256, 0, stream>>>(W_qkv, wqkt, NQKV);
  }
  {
    int n = CC * CC;
    transpose_w_to_bf16<<<(n + 255) / 256, 256, 0, stream>>>(W_out, wot, CC);
  }

  // 2) QKV projection (WMMA GEMM), scatter into attention-friendly layouts
  {
    dim3 grid(MTOT / 64, NQKV / 256);   // (128, 12)
    gemm_qkv<<<grid, 256, 0, stream>>>(xb, wqkt, Qb, Kb, Vt);
  }

  // 3) causal flash attention (WMMA QK^T and PV)
  {
    dim3 grid(TT / 64, BB * HH);        // (32, 64)
    attn_fwd<<<grid, 128, 0, stream>>>(Qb, Kb, Vt, yb);
  }

  // 4) output projection (WMMA GEMM) -> f32 result
  {
    dim3 grid(MTOT / 64, CC / 256);     // (128, 4)
    gemm_out<<<grid, 256, 0, stream>>>(yb, wot, out);
  }
}
